// DeformableConv2d_87205015978255
// MI455X (gfx1250) — matrix operand order repaired and checked by reading, not compiled, not run on hardware
//
#include <hip/hip_runtime.h>
#include <math.h>

// MI455X / gfx1250, wave32. Fully WMMA-driven DCNv2.
// K ordering is kernel-point-major: kk = k9*128 + c, so every 32-wide K chunk
// sits inside one kernel point -> no div/mod in hot loops, one bilinear
// metadata fetch per chunk, per-chunk bounds handling, native packed bf16 cvt.
// (Scalar f32 sampling arithmetic on purpose: explicit float2 pairing was
// avoided — it forces v_mov_b16 register-pair shuffles.)

typedef __attribute__((ext_vector_type(16))) __bf16 v16bf;
typedef __attribute__((ext_vector_type(8)))  float  v8f;

#define CIN   128
#define COUT  128
#define HH    64
#define WW    64
#define NCH   36          // K chunks of 32 (K = 9*128 = 1152)

// Native packed f32->bf16 (backend selects v_cvt_pk_bf16_f32)
__device__ __forceinline__ unsigned pk_bf16(float a, float b) {
    union { __bf16 h[2]; unsigned u; } t;
    t.h[0] = (__bf16)a;
    t.h[1] = (__bf16)b;
    return t.u;
}
__device__ __forceinline__ unsigned short f2bf1(float a) {
    union { __bf16 h; unsigned short u; } t;
    t.h = (__bf16)a;
    return t.u;
}

struct Meta { int a00, a01, a10, a11; float w00, w01, w10, w11; };
union AFrag { unsigned u[8]; v16bf v; };
union BFrag { uint4 q[2];   v16bf v; };

// ---------------- k0a: Wp[kc][n][k] = bf16(weight[n][c][k9]), kk=kc*32+k, k9=kk>>7, c=kk&127
__global__ void prep_weight(const float* __restrict__ w, unsigned short* __restrict__ Wp) {
    int idx = blockIdx.x * 256 + threadIdx.x;           // 147456
    if (idx >= NCH * COUT * 32) return;
    int k  = idx & 31;
    int n  = (idx >> 5) & 127;
    int kc = idx >> 12;
    int kk = kc * 32 + k;
    int k9 = kk >> 7;
    int c  = kk & 127;
    Wp[idx] = f2bf1(w[(n * CIN + c) * 9 + k9]);
}

// ---------------- k0b: Wp2[kc][n][k], n<18 offset_w, 18..26 mask_w, 27..31 zero
__global__ void prep_weight2(const float* __restrict__ ow, const float* __restrict__ mw,
                             unsigned short* __restrict__ Wp2) {
    int idx = blockIdx.x * 256 + threadIdx.x;           // 36864
    if (idx >= NCH * 32 * 32) return;
    int k  = idx & 31;
    int n  = (idx >> 5) & 31;
    int kc = idx >> 10;
    int kk = kc * 32 + k;
    int k9 = kk >> 7;
    int c  = kk & 127;
    float v = 0.f;
    if (n < 18)      v = ow[(n * CIN + c) * 9 + k9];
    else if (n < 27) v = mw[((n - 18) * CIN + c) * 9 + k9];
    Wp2[idx] = f2bf1(v);
}

// ---------------- k1: offset/mask conv on WMMA; bias + sigmoid fused in epilogue
__global__ __launch_bounds__(256)
void offset_mask_wmma(const float* __restrict__ x,
                      const unsigned short* __restrict__ Wp2,
                      const float* __restrict__ ob, const float* __restrict__ mb,
                      float* __restrict__ om) {
    const int bid = blockIdx.x;                 // 512 = (b,h) rows
    const int b = bid >> 6, h = bid & 63;
    const float* xb = x + b * CIN * HH * WW;
    const int tid = threadIdx.x, lane = tid & 31, wid = tid >> 5;
    const int mt = wid & 3, ntl = wid >> 2;     // 4 M-tiles x 2 N-tiles
    const int lm = lane & 15;
    const int m  = mt * 16 + lm;                // w coordinate
    const int hs = (lane < 16) ? 0 : 16;
    const int nb = (lane < 16) ? 0 : 16;

    v8f acc = (v8f){0.f, 0.f, 0.f, 0.f, 0.f, 0.f, 0.f, 0.f};

#pragma unroll
    for (int k9 = 0; k9 < 9; ++k9) {
        const int ki = k9 / 3, kj = k9 - 3 * ki;        // compile-time
        int yy = h + ki - 1;
        int xx = m + kj - 1;
        bool ok = ((unsigned)yy < 64u) & ((unsigned)xx < 64u);
        const float* basep = xb + (ok ? (yy * WW + xx) : 0);
        float vm = ok ? 1.f : 0.f;
#pragma unroll
        for (int cq = 0; cq < 4; ++cq) {
            const int kc = k9 * 4 + cq;
            const uint4* bp = (const uint4*)(Wp2 +
                ((size_t)(kc * 32 + ntl * 16 + lm) * 32 + nb));
            BFrag bf; bf.q[0] = bp[0]; bf.q[1] = bp[1];

            AFrag af;
            const int cb = cq * 32 + hs;
#pragma unroll
            for (int j = 0; j < 16; j += 2) {
                int c0 = cb + j;
                float v0 = basep[(size_t)c0       * (HH * WW)] * vm;
                float v1 = basep[(size_t)(c0 + 1) * (HH * WW)] * vm;
                af.u[j >> 1] = pk_bf16(v0, v1);
            }
            acc = __builtin_amdgcn_wmma_f32_16x16x32_bf16(
                false, af.v, false, bf.v, (short)0, acc, false, false);
        }
    }

    int n = ntl * 16 + lm;
    if (n < 27) {
        float bias = (n < 18) ? ob[n] : mb[n - 18];
#pragma unroll
        for (int r = 0; r < 8; ++r) {
            int pix = bid * 64 + mt * 16 + r + ((lane >> 4) << 3);
            float v = acc[r] + bias;
            if (n >= 18) v = 1.f / (1.f + __expf(-v));
            om[(size_t)pix * 28 + n] = v;
        }
    }
}

// ---------------- k2: fused deformable sampling + WMMA GEMM
union SMem {
    Meta  meta[128 * 9];        // 36864 B (main loop)
    float outs[64 * 130];       // 33280 B (epilogue, reused after barrier)
};

__global__ __launch_bounds__(256)
void deform_wmma(const float* __restrict__ x,
                 const unsigned short* __restrict__ Wp,
                 const float* __restrict__ om,
                 float* __restrict__ out) {
    __shared__ SMem sm;

    const int bid = blockIdx.x;                 // 256 blocks = 128 pixels each
    const int b  = bid >> 5;
    const int hp = bid & 31;                    // rows 2*hp .. 2*hp+1
    const int tid = threadIdx.x;
    const float* xb = x + b * CIN * HH * WW;

    // Phase 1: bilinear metadata for 128 pixels x 9 kernel points
    for (int e = tid; e < 1152; e += 256) {
        int p  = e / 9;
        int k9 = e - 9 * p;
        int ki = k9 / 3, kj = k9 - 3 * ki;
        int gp = bid * 128 + p;
        int h = (gp >> 6) & 63, w = gp & 63;
        const float* op = om + (size_t)gp * 28;
        float offy = op[2 * k9], offx = op[2 * k9 + 1], msk = op[18 + k9];
        float py = (float)(h + ki - 1) + offy;
        float px = (float)(w + kj - 1) + offx;
        float y0f = floorf(py), x0f = floorf(px);
        int   y0 = (int)y0f,   x0 = (int)x0f;
        float ly = py - y0f,   lx = px - x0f;
        int y0c = min(max(y0, 0), 63),     y1c = min(max(y0 + 1, 0), 63);
        int x0c = min(max(x0, 0), 63),     x1c = min(max(x0 + 1, 0), 63);
        float vy0 = ((unsigned)y0       < 64u) ? 1.f : 0.f;
        float vy1 = ((unsigned)(y0 + 1) < 64u) ? 1.f : 0.f;
        float vx0 = ((unsigned)x0       < 64u) ? 1.f : 0.f;
        float vx1 = ((unsigned)(x0 + 1) < 64u) ? 1.f : 0.f;
        Meta md;
        md.a00 = y0c * 64 + x0c;  md.a01 = y0c * 64 + x1c;
        md.a10 = y1c * 64 + x0c;  md.a11 = y1c * 64 + x1c;
        md.w00 = (1.f - ly) * (1.f - lx) * msk * vy0 * vx0;
        md.w01 = (1.f - ly) * lx         * msk * vy0 * vx1;
        md.w10 = ly * (1.f - lx)         * msk * vy1 * vx0;
        md.w11 = ly * lx                 * msk * vy1 * vx1;
        sm.meta[e] = md;
    }
    __syncthreads();

    // Phase 2: each wave = one 16-pixel M-tile x all 128 output channels
    const int lane = tid & 31;
    const int wid  = tid >> 5;
    const int lm   = lane & 15;
    const int prow = wid * 16 + lm;             // pixel index 0..127
    const Meta* mrb = &sm.meta[prow * 9];
    const int hs = (lane < 16) ? 0 : 16;
    const int nb = (lane < 16) ? 0 : 16;

    v8f acc[8];
#pragma unroll
    for (int nt = 0; nt < 8; ++nt)
        acc[nt] = (v8f){0.f, 0.f, 0.f, 0.f, 0.f, 0.f, 0.f, 0.f};

    for (int k9 = 0; k9 < 9; ++k9) {
        Meta md = mrb[k9];                      // ONE 32B meta read per 4 chunks
#pragma unroll
        for (int cq = 0; cq < 4; ++cq) {
            const int kc = k9 * 4 + cq;
            if (kc + 1 < NCH)
                __builtin_prefetch(Wp + (size_t)(kc + 1) * (COUT * 32) + lane * 128, 0, 0);

            // B fragments first: 16 b128 loads in flight during the A build
            BFrag bf[8];
#pragma unroll
            for (int nt = 0; nt < 8; ++nt) {
                const uint4* bp = (const uint4*)(Wp +
                    ((size_t)(kc * COUT + nt * 16 + lm) * 32 + nb));
                bf[nt].q[0] = bp[0];
                bf[nt].q[1] = bp[1];
            }

            // A fragment: 16 bilinear samples, shared corner offsets/weights
            AFrag af;
            const int cb = cq * 32 + hs;
#pragma unroll
            for (int j = 0; j < 16; j += 2) {
                int c0 = cb + j;
                const float* pl0 = xb + (size_t)c0       * (HH * WW);
                const float* pl1 = xb + (size_t)(c0 + 1) * (HH * WW);
                float v0 = md.w00 * pl0[md.a00] + md.w01 * pl0[md.a01]
                         + md.w10 * pl0[md.a10] + md.w11 * pl0[md.a11];
                float v1 = md.w00 * pl1[md.a00] + md.w01 * pl1[md.a01]
                         + md.w10 * pl1[md.a10] + md.w11 * pl1[md.a11];
                af.u[j >> 1] = pk_bf16(v0, v1);
            }

#pragma unroll
            for (int nt = 0; nt < 8; ++nt)
                acc[nt] = __builtin_amdgcn_wmma_f32_16x16x32_bf16(
                    false, af.v, false, bf[nt].v, (short)0, acc[nt], false, false);
        }
    }

    __syncthreads();            // metadata dead; LDS reused for transpose
    // Epilogue: two channel-halves through padded LDS, coalesced b128 stores
#pragma unroll
    for (int ph = 0; ph < 2; ++ph) {
#pragma unroll
        for (int q = 0; q < 4; ++q) {
            int nt = ph * 4 + q;
            int ol = q * 16 + lm;                       // local channel 0..63
            int pc = wid * 16 + ((lane >> 4) << 3);     // pixel base
#pragma unroll
            for (int r = 0; r < 8; ++r)
                sm.outs[ol * 130 + pc + r] = acc[nt][r];
        }
        __syncthreads();
        int ol  = tid >> 2;
        int seg = (tid & 3) * 32;
        int o   = ph * 64 + ol;
        float* op2 = out + ((size_t)(b * COUT + o) * (HH * WW) + hp * 128 + seg);
#pragma unroll
        for (int i = 0; i < 32; i += 4) {
            float4 v4;
            v4.x = sm.outs[ol * 130 + seg + i + 0];
            v4.y = sm.outs[ol * 130 + seg + i + 1];
            v4.z = sm.outs[ol * 130 + seg + i + 2];
            v4.w = sm.outs[ol * 130 + seg + i + 3];
            *(float4*)(op2 + i) = v4;
        }
        __syncthreads();
    }
}

extern "C" void kernel_launch(void* const* d_in, const int* in_sizes, int n_in,
                              void* d_out, int out_size, void* d_ws, size_t ws_size,
                              hipStream_t stream) {
    const float* x  = (const float*)d_in[0];
    const float* w  = (const float*)d_in[1];
    const float* ow = (const float*)d_in[2];
    const float* ob = (const float*)d_in[3];
    const float* mw = (const float*)d_in[4];
    const float* mb = (const float*)d_in[5];
    float* out = (float*)d_out;

    unsigned short* Wp  = (unsigned short*)d_ws;                       // 294912 B
    unsigned short* Wp2 = (unsigned short*)((char*)d_ws + 294912);     //  73728 B
    float*          om  = (float*)((char*)d_ws + 294912 + 73728);      // 32768*28*4 B

    prep_weight <<<576, 256, 0, stream>>>(w, Wp);
    prep_weight2<<<144, 256, 0, stream>>>(ow, mw, Wp2);
    offset_mask_wmma<<<512, 256, 0, stream>>>(x, Wp2, ob, mb, om);
    deform_wmma<<<256, 256, 0, stream>>>(x, Wp, om, out);
}
